// FusionAttentionBlock_14851996910239
// MI455X (gfx1250) — matrix operand order repaired and checked by reading, not compiled, not run on hardware
//
#include <hip/hip_runtime.h>

#define Dq   128
#define BQ   524288
#define EPSQ 1e-5f

typedef _Float16 v16h __attribute__((ext_vector_type(16)));
typedef _Float16 v8h  __attribute__((ext_vector_type(8)));
typedef _Float16 v4h  __attribute__((ext_vector_type(4)));
typedef float    v8f  __attribute__((ext_vector_type(8)));
typedef float    v4f  __attribute__((ext_vector_type(4)));

__device__ __forceinline__ v16h cat16(v8h a, v8h b) {
  return __builtin_shufflevector(a, b, 0,1,2,3,4,5,6,7,8,9,10,11,12,13,14,15);
}
__device__ __forceinline__ v8f wmma16(v16h a, v16h b, v8f c) {
  return __builtin_amdgcn_wmma_f32_16x16x32_f16(false, a, false, b, (short)0, c, false, false);
}

// ---------------------------------------------------------------------------
// Kernel 0: zero the Gram/sum accumulators.
// ---------------------------------------------------------------------------
__global__ void zero_kernel(float* p, int n) {
  int i = blockIdx.x * 256 + threadIdx.x;
  if (i < n) p[i] = 0.f;
}

// ---------------------------------------------------------------------------
// Kernel 1: Gram matrices G11=x1'x1 (p=0), G12=x1'x2 (p=1), G22=x2'x2 (p=2),
// plus column sums s1,s2. K=batch streamed in chunks of 32; f16 hi/lo split
// WMMA with f32 accumulation; partial 128x128 tiles atomically reduced.
// ---------------------------------------------------------------------------
#define GP     40           // LDS pitch in halfs per feature row (32 data + 8 pad) = 80B
#define NCHUNK (BQ / 32)

__global__ __launch_bounds__(256) void gram_kernel(
    const float* __restrict__ x1, const float* __restrict__ x2,
    float* __restrict__ G, float* __restrict__ s1, float* __restrict__ s2) {
  __shared__ _Float16 Ah[128 * GP], Al[128 * GP], Bh2[128 * GP], Bl2[128 * GP];
  const int p = blockIdx.y;
  const float* xa = (p == 2) ? x2 : x1;
  const float* xb = (p == 0) ? x1 : x2;
  _Float16* Bh = (p == 1) ? Bh2 : Ah;
  _Float16* Bl = (p == 1) ? Bl2 : Al;

  const int t = threadIdx.x;
  const int f = t & 127, kh = t >> 7;             // fill mapping: feature f, k-half kh
  const int w = t >> 5, lane = t & 31;            // compute mapping
  const int m = lane & 15, hl = lane >> 4;

  v8f zero = {};
  v8f acc[8];
#pragma unroll
  for (int n = 0; n < 8; ++n) acc[n] = zero;
  float sAcc = 0.f;

  for (int c = blockIdx.x; c < NCHUNK; c += gridDim.x) {
    int cn = c + gridDim.x;
    if (cn < NCHUNK)  // prefetch next chunk (global_prefetch_b8)
      __builtin_prefetch(xa + (size_t)(cn * 32 + kh * 16) * Dq + f, 0, 1);

    // ---- fill LDS (transposed, f16 hi/lo split) ----
    {
      const float* src = xa + (size_t)(c * 32 + kh * 16) * Dq + f;
      float v[16];
#pragma unroll
      for (int i = 0; i < 16; ++i) v[i] = src[(size_t)i * Dq];
      if (p != 1) {
#pragma unroll
        for (int i = 0; i < 16; ++i) sAcc += v[i];
      }
#pragma unroll
      for (int g = 0; g < 4; ++g) {
        v4h hi, lo;
#pragma unroll
        for (int e = 0; e < 4; ++e) {
          float xx = v[g * 4 + e];
          _Float16 h = (_Float16)xx;
          hi[e] = h; lo[e] = (_Float16)(xx - (float)h);
        }
        *(v4h*)&Ah[f * GP + kh * 16 + g * 4] = hi;
        *(v4h*)&Al[f * GP + kh * 16 + g * 4] = lo;
      }
      if (p == 1) {
        const float* srcb = xb + (size_t)(c * 32 + kh * 16) * Dq + f;
        float vb[16];
#pragma unroll
        for (int i = 0; i < 16; ++i) vb[i] = srcb[(size_t)i * Dq];
#pragma unroll
        for (int g = 0; g < 4; ++g) {
          v4h hi, lo;
#pragma unroll
          for (int e = 0; e < 4; ++e) {
            float xx = vb[g * 4 + e];
            _Float16 h = (_Float16)xx;
            hi[e] = h; lo[e] = (_Float16)(xx - (float)h);
          }
          *(v4h*)&Bh2[f * GP + kh * 16 + g * 4] = hi;
          *(v4h*)&Bl2[f * GP + kh * 16 + g * 4] = lo;
        }
      }
    }
    __syncthreads();

    // ---- WMMA: wave w owns output rows [16w,16w+16), all 8 column tiles ----
    {
      const _Float16* apH = &Ah[(16 * w + m) * GP + hl * 8];
      const _Float16* apL = &Al[(16 * w + m) * GP + hl * 8];
      v16h aH = cat16(*(const v8h*)apH, *(const v8h*)(apH + 16));
      v16h aL = cat16(*(const v8h*)apL, *(const v8h*)(apL + 16));
#pragma unroll
      for (int n = 0; n < 8; ++n) {
        const _Float16* bpH = &Bh[(16 * n + m) * GP + hl * 8];
        const _Float16* bpL = &Bl[(16 * n + m) * GP + hl * 8];
        v16h bH = cat16(*(const v8h*)bpH, *(const v8h*)(bpH + 16));
        v16h bL = cat16(*(const v8h*)bpL, *(const v8h*)(bpL + 16));
        acc[n] = wmma16(aH, bH, acc[n]);   // hi*hi
        acc[n] = wmma16(aH, bL, acc[n]);   // hi*lo
        acc[n] = wmma16(aL, bH, acc[n]);   // lo*hi
      }
    }
    __syncthreads();
  }

  // ---- reduce partials into global Gram ----
  float* Gp = G + p * (Dq * Dq);
#pragma unroll
  for (int n = 0; n < 8; ++n)
#pragma unroll
    for (int r = 0; r < 8; ++r)
      atomicAdd(&Gp[(16 * w + r + hl * 8) * Dq + 16 * n + m], acc[n][r]);
  if (p == 0) atomicAdd(&s1[f], sAcc);
  if (p == 2) atomicAdd(&s2[f], sAcc);
}

// ---------------------------------------------------------------------------
// Kernel 2: all the tiny [128x128] algebra on one WGP (32 waves):
// S -> softmax -> A=attn*Wv', C = I + W'A (BN scale folded), d vectors.
// ---------------------------------------------------------------------------
#define P2 129

__global__ __launch_bounds__(1024) void prep_kernel(
    const float* __restrict__ Wq, const float* __restrict__ bq,
    const float* __restrict__ Wk, const float* __restrict__ bk,
    const float* __restrict__ Wv1, const float* __restrict__ bv1,
    const float* __restrict__ Wv2, const float* __restrict__ bv2,
    const float* __restrict__ gamma, const float* __restrict__ beta,
    const float* __restrict__ G, const float* __restrict__ s1,
    const float* __restrict__ s2, float* __restrict__ attn_ws,
    float* __restrict__ d1, float* __restrict__ d2,
    _Float16* __restrict__ C1hT, _Float16* __restrict__ C1lT,
    _Float16* __restrict__ C2hT, _Float16* __restrict__ C2lT) {
  __shared__ float bw[Dq * P2];
  __shared__ float bm[Dq * P2];
  __shared__ float u1v[Dq], u2v[Dq], cvec[Dq], scale_s[Dq], red[8 * Dq];

  const float* G11 = G;
  const float* G12 = G + Dq * Dq;
  const float* G22 = G + 2 * Dq * Dq;
  const int t = threadIdx.x;
  const int j = t & 127, ib = t >> 7, i0 = ib * 16;
  const float Bf = (float)BQ;
  float r[16];

  auto loadN = [&](float* dst, const float* src) {
    for (int idx = t; idx < Dq * Dq; idx += 1024)
      dst[(idx >> 7) * P2 + (idx & 127)] = src[idx];
  };
  auto loadT = [&](float* dst, const float* src) {
    for (int idx = t; idx < Dq * Dq; idx += 1024)
      dst[(idx & 127) * P2 + (idx >> 7)] = src[idx];
  };
  auto mmNN = [&](const float* A, const float* Bm) {  // r = A * Bm
#pragma unroll
    for (int x = 0; x < 16; ++x) r[x] = 0.f;
    for (int k = 0; k < Dq; ++k) {
      float bvv = Bm[k * P2 + j];
#pragma unroll
      for (int x = 0; x < 16; ++x) r[x] += A[(i0 + x) * P2 + k] * bvv;
    }
  };
  auto mmNT = [&](const float* A, const float* Bm) {  // r = A * Bm^T
#pragma unroll
    for (int x = 0; x < 16; ++x) r[x] = 0.f;
    for (int k = 0; k < Dq; ++k) {
      float bvv = Bm[j * P2 + k];
#pragma unroll
      for (int x = 0; x < 16; ++x) r[x] += A[(i0 + x) * P2 + k] * bvv;
    }
  };
  auto dump = [&](float* dst) {
#pragma unroll
    for (int x = 0; x < 16; ++x) dst[(i0 + x) * P2 + j] = r[x];
  };

  // ---- step 1: T1 = Wq * G12 ; u1 = Wq*s1 ----
  loadN(bw, Wq); loadN(bm, G12);
  __syncthreads();
  if (t < 128) { float a = 0; for (int k = 0; k < Dq; ++k) a += bw[t * P2 + k] * s1[k]; u1v[t] = a; }
  mmNN(bw, bm);
  __syncthreads();
  dump(bm);                       // bm = T1
  __syncthreads();

  // ---- step 2: S = T1*Wk' + u1*bk' + bq*u2' + B*bq*bk' ----
  loadN(bw, Wk);
  __syncthreads();
  if (t < 128) { float a = 0; for (int k = 0; k < Dq; ++k) a += bw[t * P2 + k] * s2[k]; u2v[t] = a; }
  mmNT(bm, bw);
  __syncthreads();
#pragma unroll
  for (int x = 0; x < 16; ++x)
    r[x] += u1v[i0 + x] * bk[j] + bq[i0 + x] * u2v[j] + Bf * bq[i0 + x] * bk[j];
  dump(bm);                       // bm = S
  __syncthreads();

  // ---- step 3: row softmax ----
  if (t < 128) {
    float mx = -1e30f;
    for (int c = 0; c < Dq; ++c) mx = fmaxf(mx, bm[t * P2 + c]);
    float sm = 0.f;
    for (int c = 0; c < Dq; ++c) { float e = expf(bm[t * P2 + c] - mx); bm[t * P2 + c] = e; sm += e; }
    float inv = 1.0f / sm;
    for (int c = 0; c < Dq; ++c) bm[t * P2 + c] *= inv;
  }
  __syncthreads();
  for (int idx = t; idx < Dq * Dq; idx += 1024)
    attn_ws[idx] = bm[(idx >> 7) * P2 + (idx & 127)];
  __syncthreads();

  // ---- two symmetric branches ----
  for (int br = 0; br < 2; ++br) {
    const float* Wx = br ? Wk : Wq;   const float* bx = br ? bk : bq;
    const float* Wv = br ? Wv2 : Wv1; const float* bv = br ? bv2 : bv1;
    const float* Gx = br ? G22 : G11; const float* sx = br ? s2 : s1;
    float* dx = br ? d2 : d1;
    _Float16* ChT = br ? C2hT : C1hT;
    _Float16* ClT = br ? C2lT : C1lT;

    if (br == 1) {                       // reload attn
      for (int idx = t; idx < Dq * Dq; idx += 1024)
        bm[(idx >> 7) * P2 + (idx & 127)] = attn_ws[idx];
    }
    loadN(bw, Wv);
    __syncthreads();
    mmNT(bm, bw);                        // A_ = attn * Wv'
    __syncthreads();
    dump(bm);                            // bm = A_
    __syncthreads();
    if (t < 128) {                       // c = bx @ A_ + bv
      float a = 0; for (int k = 0; k < Dq; ++k) a += bx[k] * bm[k * P2 + t];
      cvec[t] = a + bv[t];
    }
    loadT(bw, Wx);                       // bw = Wx'
    __syncthreads();
    mmNN(bw, bm);                        // C = Wx' * A_
#pragma unroll
    for (int x = 0; x < 16; ++x) if (i0 + x == j) r[x] += 1.0f;   // + I
    __syncthreads();
    dump(bm);                            // bm = C
    __syncthreads();
    loadN(bw, Gx);
    __syncthreads();
    mmNN(bw, bm);                        // t1 = Gx * C
    {
      float pd = 0.f;
#pragma unroll
      for (int x = 0; x < 16; ++x) pd += bm[(i0 + x) * P2 + j] * r[x];
      red[ib * Dq + j] = pd;             // partial of (C' Gx C)_jj
    }
    __syncthreads();
    if (t < 128) {
      float quad = 0.f;
      for (int g = 0; g < 8; ++g) quad += red[g * Dq + t];
      float scj = 0.f;
      for (int k = 0; k < Dq; ++k) scj += sx[k] * bm[k * P2 + t];  // (s'C)_j
      float cj = cvec[t];
      float mu = scj / Bf + cj;
      float m2 = (quad + 2.f * cj * scj) / Bf + cj * cj;
      float var = m2 - mu * mu;
      float sc = gamma[t] * rsqrtf(var + EPSQ);
      scale_s[t] = sc;
      dx[t] = (cj - mu) * sc + beta[t];
    }
    __syncthreads();
    for (int idx = t; idx < Dq * Dq; idx += 1024) {  // write C*scale, transposed, hi/lo
      int fi = idx >> 7, jj = idx & 127;
      float v = bm[fi * P2 + jj] * scale_s[jj];
      _Float16 h = (_Float16)v;
      ChT[jj * Dq + fi] = h;
      ClT[jj * Dq + fi] = (_Float16)(v - (float)h);
    }
    __syncthreads();
  }
}

// ---------------------------------------------------------------------------
// Kernel 3: y = x @ Cs + d   (split-f16 WMMA, streaming, C cached in LDS)
// ---------------------------------------------------------------------------
#define PO 136   // LDS pitch in halfs (128 + 8 pad) = 272B

__global__ __launch_bounds__(256) void out_kernel(
    const float* __restrict__ x1, const float* __restrict__ x2,
    const _Float16* __restrict__ C1hT, const _Float16* __restrict__ C1lT,
    const _Float16* __restrict__ C2hT, const _Float16* __restrict__ C2lT,
    const float* __restrict__ d1, const float* __restrict__ d2,
    float* __restrict__ out) {
  __shared__ _Float16 xh[128 * PO], xl[128 * PO], ch[128 * PO], cl[128 * PO];
  __shared__ float dsh[Dq];

  const int br = blockIdx.y;
  const float* x = br ? x2 : x1;
  const _Float16* Ch = br ? C2hT : C1hT;
  const _Float16* Cl = br ? C2lT : C1lT;
  const float* dv = br ? d2 : d1;
  float* yp = out + (size_t)br * BQ * Dq;

  const int t = threadIdx.x;
  // load Cs (hi/lo, already transposed [j][f]) once per block
  for (int idx = t; idx < 2048; idx += 256) {
    int jj = idx >> 4, c8 = idx & 15;
    *(v8h*)&ch[jj * PO + c8 * 8] = ((const v8h*)Ch)[idx];
    *(v8h*)&cl[jj * PO + c8 * 8] = ((const v8h*)Cl)[idx];
  }
  if (t < 128) dsh[t] = dv[t];

  const int w = t >> 5, lane = t & 31, m = lane & 15, hl = lane >> 4;
  const int nch = BQ / 128;
  v8f zero = {};

  for (int rc = blockIdx.x; rc < nch; rc += gridDim.x) {
    int rn = rc + gridDim.x;
    if (rn < nch)
      __builtin_prefetch(x + (size_t)rn * 128 * Dq + t * 64, 0, 1);
    __syncthreads();
    // fill x chunk (128 rows), converted to f16 hi/lo
    for (int idx = t; idx < 4096; idx += 256) {
      int row = idx >> 5, c4 = idx & 31;
      v4f v = ((const v4f*)(x + (size_t)rc * 128 * Dq))[idx];
      v4h hi, lo;
#pragma unroll
      for (int e = 0; e < 4; ++e) {
        float xx = v[e];
        _Float16 h = (_Float16)xx;
        hi[e] = h; lo[e] = (_Float16)(xx - (float)h);
      }
      *(v4h*)&xh[row * PO + c4 * 4] = hi;
      *(v4h*)&xl[row * PO + c4 * 4] = lo;
    }
    __syncthreads();

    v8f acc[8];
#pragma unroll
    for (int n = 0; n < 8; ++n) acc[n] = zero;
#pragma unroll
    for (int kc = 0; kc < 4; ++kc) {
      const _Float16* apH = &xh[(16 * w + m) * PO + kc * 32 + hl * 8];
      const _Float16* apL = &xl[(16 * w + m) * PO + kc * 32 + hl * 8];
      v16h aH = cat16(*(const v8h*)apH, *(const v8h*)(apH + 16));
      v16h aL = cat16(*(const v8h*)apL, *(const v8h*)(apL + 16));
#pragma unroll
      for (int n = 0; n < 8; ++n) {
        const _Float16* bpH = &ch[(16 * n + m) * PO + kc * 32 + hl * 8];
        const _Float16* bpL = &cl[(16 * n + m) * PO + kc * 32 + hl * 8];
        v16h bH = cat16(*(const v8h*)bpH, *(const v8h*)(bpH + 16));
        v16h bL = cat16(*(const v8h*)bpL, *(const v8h*)(bpL + 16));
        acc[n] = wmma16(aH, bH, acc[n]);
        acc[n] = wmma16(aH, bL, acc[n]);
        acc[n] = wmma16(aL, bH, acc[n]);
      }
    }
    float* o = yp + (size_t)rc * 128 * Dq;
#pragma unroll
    for (int n = 0; n < 8; ++n)
#pragma unroll
      for (int rr = 0; rr < 8; ++rr)
        o[(16 * w + rr + hl * 8) * Dq + 16 * n + m] = acc[n][rr] + dsh[16 * n + m];
  }
}

// ---------------------------------------------------------------------------
extern "C" void kernel_launch(void* const* d_in, const int* in_sizes, int n_in,
                              void* d_out, int out_size, void* d_ws, size_t ws_size,
                              hipStream_t stream) {
  const float* x1   = (const float*)d_in[0];
  const float* x2   = (const float*)d_in[1];
  const float* Wq   = (const float*)d_in[2];
  const float* bq   = (const float*)d_in[3];
  const float* Wk   = (const float*)d_in[4];
  const float* bk   = (const float*)d_in[5];
  const float* Wv1  = (const float*)d_in[6];
  const float* bv1  = (const float*)d_in[7];
  const float* Wv2  = (const float*)d_in[8];
  const float* bv2  = (const float*)d_in[9];
  const float* gam  = (const float*)d_in[10];
  const float* bet  = (const float*)d_in[11];
  float* out = (float*)d_out;

  // workspace layout (~390 KB total)
  float* ws   = (float*)d_ws;
  float* G    = ws;                   // 3 * 16384
  float* s1   = ws + 49152;           // 128
  float* s2   = ws + 49280;           // 128
  float* attn = ws + 49408;           // 16384
  float* d1   = ws + 65792;           // 128
  float* d2   = ws + 65920;           // 128
  _Float16* Chalf = (_Float16*)(ws + 66048);
  _Float16* C1hT = Chalf;
  _Float16* C1lT = Chalf + 16384;
  _Float16* C2hT = Chalf + 32768;
  _Float16* C2lT = Chalf + 49152;

  zero_kernel<<<(49408 + 255) / 256, 256, 0, stream>>>(G, 49408);
  gram_kernel<<<dim3(1024, 3), 256, 0, stream>>>(x1, x2, G, s1, s2);
  prep_kernel<<<1, 1024, 0, stream>>>(Wq, bq, Wk, bk, Wv1, bv1, Wv2, bv2,
                                      gam, bet, G, s1, s2, attn, d1, d2,
                                      C1hT, C1lT, C2hT, C2lT);
  out_kernel<<<dim3(2048, 2), 256, 0, stream>>>(x1, x2, C1hT, C1lT, C2hT, C2lT,
                                                d1, d2, out);
}
